// MLayer_47064251630009
// MI455X (gfx1250) — matrix operand order repaired and checked by reading, not compiled, not run on hardware
//
#include <hip/hip_runtime.h>

#define DIM 1024
#define H   8
#define HD  128
#define R   8
#define K1  32   // HD/4

#define BM 128
#define BN 128
#define BK 32
#define NK (DIM / BK)          // 32 k-steps
#define LPITCH 80              // 64B row + 16B pad (5 x 16B -> conflict-free b128 reads)
#define TILE_BYTES (BM * LPITCH)   // 10240 B per tile buffer

typedef __attribute__((ext_vector_type(16))) _Float16 v16h;
typedef __attribute__((ext_vector_type(8)))  float    v8f;
typedef int v4i __attribute__((vector_size(16)));
typedef __attribute__((address_space(1))) v4i* gptr_v4i;
typedef __attribute__((address_space(3))) v4i* lptr_v4i;
typedef __attribute__((address_space(3))) char* lchar;
typedef __attribute__((address_space(3))) const v4i* lcv4;
typedef __attribute__((address_space(3))) v4i* lv4;

#ifndef __has_builtin
#define __has_builtin(x) 0
#endif

#if __has_builtin(__builtin_amdgcn_global_load_async_to_lds_b128)
#define HAVE_ASYNC_LDS 1
#else
#define HAVE_ASYNC_LDS 0
#endif

__device__ __forceinline__ void async_copy16(lchar lds_dst, const void* gsrc)
{
#if HAVE_ASYNC_LDS
    __builtin_amdgcn_global_load_async_to_lds_b128(
        (gptr_v4i)(v4i*)gsrc, (lptr_v4i)lds_dst, 0, 0);
#else
    *(lv4)lds_dst = *(const v4i*)gsrc;   // fallback: synchronous LDS staging
#endif
}

__device__ __forceinline__ void wait_async()
{
#if HAVE_ASYNC_LDS
#if __has_builtin(__builtin_amdgcn_s_wait_asynccnt)
    __builtin_amdgcn_s_wait_asynccnt(0);
#else
    asm volatile("s_wait_asynccnt 0x0" ::: "memory");
#endif
#endif
}

__device__ __forceinline__ float wred_sum(float v) {
#pragma unroll
    for (int o = 16; o > 0; o >>= 1) v += __shfl_xor(v, o, 32);
    return v;
}

// ---------------------------------------------------------------------------
// Physics: LayerNorm + gating MLP + low-rank Christoffel + Heun step.
// One block (8 waves) per row; wave w = head w; 4 f32 per lane (wave32, HD=128).
// ---------------------------------------------------------------------------
__device__ __forceinline__ float4 christoffel(const float4 vvv, const float4 xx,
                                              const float* __restrict__ Wx,
                                              const float* __restrict__ Wv,
                                              const float* __restrict__ Wo,
                                              int lane)
{
    const float* wx = Wx + lane * 4 * R;
    const float* wv = Wv + lane * 4 * R;
    float hp[R], qp[R];
#pragma unroll
    for (int r = 0; r < R; ++r) { hp[r] = 0.f; qp[r] = 0.f; }
    const float xs[4] = {xx.x, xx.y, xx.z, xx.w};
    const float vs[4] = {vvv.x, vvv.y, vvv.z, vvv.w};
#pragma unroll
    for (int i = 0; i < 4; ++i) {
#pragma unroll
        for (int r = 0; r < R; ++r) {
            hp[r] = fmaf(xs[i], wx[i * R + r], hp[r]);
            qp[r] = fmaf(vs[i], wv[i * R + r], qp[r]);
        }
    }
    float coef[R];
#pragma unroll
    for (int r = 0; r < R; ++r) {
        float hr = tanhf(wred_sum(hp[r]));
        float qr = wred_sum(qp[r]);
        coef[r] = hr * qr * qr;
    }
    float4 o = {0.f, 0.f, 0.f, 0.f};
#pragma unroll
    for (int r = 0; r < R; ++r) {
        float4 wo = *(const float4*)(Wo + r * HD + lane * 4);
        o.x = fmaf(coef[r], wo.x, o.x);
        o.y = fmaf(coef[r], wo.y, o.y);
        o.z = fmaf(coef[r], wo.z, o.z);
        o.w = fmaf(coef[r], wo.w, o.w);
    }
    return o;
}

__global__ __launch_bounds__(256)
void physics_kernel(const float* __restrict__ x, const float* __restrict__ v,
                    const float* __restrict__ force,
                    const float* __restrict__ ln_g, const float* __restrict__ ln_b,
                    const float* __restrict__ gW1, const float* __restrict__ gb1,
                    const float* __restrict__ gW2, const float* __restrict__ gb2,
                    const float* __restrict__ dtp,
                    const float* __restrict__ cWx, const float* __restrict__ cWv,
                    const float* __restrict__ cWo,
                    _Float16* __restrict__ Xcat, _Float16* __restrict__ Vcat)
{
    const int row   = blockIdx.x;
    const int w     = threadIdx.x >> 5;
    const int lane  = threadIdx.x & 31;
    const int dbase = w * HD + lane * 4;
    const size_t gbase = (size_t)row * DIM + dbase;

    float4 xv = *(const float4*)(x + gbase);
    float4 vv = *(const float4*)(v + gbase);
    float4 fv = *(const float4*)(force + gbase);

    // LayerNorm over the full 1024-dim row (cross-wave via LDS)
    __shared__ float s_sum[H], s_sq[H];
    float ps = xv.x + xv.y + xv.z + xv.w;
    float pq = xv.x * xv.x + xv.y * xv.y + xv.z * xv.z + xv.w * xv.w;
    ps = wred_sum(ps); pq = wred_sum(pq);
    if (lane == 0) { s_sum[w] = ps; s_sq[w] = pq; }
    __syncthreads();
    float ts = 0.f, tq = 0.f;
#pragma unroll
    for (int i = 0; i < H; ++i) { ts += s_sum[i]; tq += s_sq[i]; }
    const float mu   = ts * (1.0f / DIM);
    const float var  = tq * (1.0f / DIM) - mu * mu;
    const float rstd = rsqrtf(var + 1e-5f);
    float4 gg = *(const float4*)(ln_g + dbase);
    float4 bb = *(const float4*)(ln_b + dbase);
    float4 xn;
    xn.x = (xv.x - mu) * rstd * gg.x + bb.x;
    xn.y = (xv.y - mu) * rstd * gg.y + bb.y;
    xn.z = (xv.z - mu) * rstd * gg.z + bb.z;
    xn.w = (xv.w - mu) * rstd * gg.w + bb.w;

    // Gating: (128->32) tanh (32->1) sigmoid; lane == hidden index k
    const float* W1 = gW1 + (size_t)w * HD * K1;
    float acc = 0.f;
    for (int sl = 0; sl < 32; ++sl) {
        float b0 = __shfl(xn.x, sl, 32);
        float b1 = __shfl(xn.y, sl, 32);
        float b2 = __shfl(xn.z, sl, 32);
        float b3 = __shfl(xn.w, sl, 32);
        const float* Wr = W1 + (sl * 4) * K1 + lane;
        acc = fmaf(b0, Wr[0 * K1], acc);
        acc = fmaf(b1, Wr[1 * K1], acc);
        acc = fmaf(b2, Wr[2 * K1], acc);
        acc = fmaf(b3, Wr[3 * K1], acc);
    }
    float h1   = tanhf(acc + gb1[w * K1 + lane]);
    float gpre = wred_sum(h1 * gW2[w * K1 + lane]) + gb2[w];
    float gate = 1.0f / (1.0f + expf(-gpre));
    float sp   = log1pf(expf(dtp[w]));            // softplus
    float dt   = 0.1f * (sp * gate / 0.1f);       // DT * scale

    float nv2 = wred_sum(vv.x * vv.x + vv.y * vv.y + vv.z * vv.z + vv.w * vv.w);
    float inv = 1.0f / (sqrtf(nv2) + 1e-6f);
    float4 vu = {vv.x * inv, vv.y * inv, vv.z * inv, vv.w * inv};

    const float* Wx = cWx + (size_t)w * HD * R;
    const float* Wv = cWv + (size_t)w * HD * R;
    const float* Wo = cWo + (size_t)w * R * HD;

    float4 c1 = christoffel(vu, xn, Wx, Wv, Wo, lane);
    float4 a1 = {-c1.x + fv.x - 0.05f * vu.x, -c1.y + fv.y - 0.05f * vu.y,
                 -c1.z + fv.z - 0.05f * vu.z, -c1.w + fv.w - 0.05f * vu.w};
    float4 xe = {xn.x + dt * vu.x, xn.y + dt * vu.y, xn.z + dt * vu.z, xn.w + dt * vu.w};
    float4 ve = {vu.x + dt * a1.x, vu.y + dt * a1.y, vu.z + dt * a1.z, vu.w + dt * a1.w};
    float4 c2 = christoffel(ve, xe, Wx, Wv, Wo, lane);
    float4 a2 = {-c2.x + fv.x - 0.05f * ve.x, -c2.y + fv.y - 0.05f * ve.y,
                 -c2.z + fv.z - 0.05f * ve.z, -c2.w + fv.w - 0.05f * ve.w};
    float4 xo = {xn.x + 0.5f * dt * (vu.x + ve.x), xn.y + 0.5f * dt * (vu.y + ve.y),
                 xn.z + 0.5f * dt * (vu.z + ve.z), xn.w + 0.5f * dt * (vu.w + ve.w)};
    float4 vo = {vu.x + 0.5f * dt * (a1.x + a2.x), vu.y + 0.5f * dt * (a1.y + a2.y),
                 vu.z + 0.5f * dt * (a1.z + a2.z), vu.w + 0.5f * dt * (a1.w + a2.w)};
    float no2 = wred_sum(vo.x * vo.x + vo.y * vo.y + vo.z * vo.z + vo.w * vo.w);
    float invn = 1.0f / (sqrtf(no2) + 1e-6f);
    vo.x *= invn; vo.y *= invn; vo.z *= invn; vo.w *= invn;

    union { _Float16 h[4]; uint2 u; } pxo, pvo;
    pxo.h[0] = (_Float16)xo.x; pxo.h[1] = (_Float16)xo.y;
    pxo.h[2] = (_Float16)xo.z; pxo.h[3] = (_Float16)xo.w;
    pvo.h[0] = (_Float16)vo.x; pvo.h[1] = (_Float16)vo.y;
    pvo.h[2] = (_Float16)vo.z; pvo.h[3] = (_Float16)vo.w;
    *(uint2*)(Xcat + gbase) = pxo.u;
    *(uint2*)(Vcat + gbase) = pvo.u;
}

// ---------------------------------------------------------------------------
// f32 KxN weight -> f16 NxK (transposed) via LDS tile.
// ---------------------------------------------------------------------------
__global__ __launch_bounds__(256)
void convert_w(const float* __restrict__ W, _Float16* __restrict__ Wt)
{
    __shared__ float tile[32][33];
    const int tx = threadIdx.x;
    const int ty = threadIdx.y;
    const int n0 = blockIdx.x * 32;
    const int k0 = blockIdx.y * 32;
#pragma unroll
    for (int j = 0; j < 4; ++j)
        tile[ty + 8 * j][tx] = W[(size_t)(k0 + ty + 8 * j) * DIM + n0 + tx];
    __syncthreads();
#pragma unroll
    for (int j = 0; j < 4; ++j)
        Wt[(size_t)(n0 + ty + 8 * j) * DIM + k0 + tx] = (_Float16)tile[tx][ty + 8 * j];
}

// ---------------------------------------------------------------------------
// WMMA GEMM: C[m,n] = sum_k A[m,k] * Wt[n,k] (+ bias + optional residual).
// 128x128 block tile, 8 waves each 64x32 (4x2 WMMA tiles), K-step 32.
// A/B tiles staged in LDS via async global->LDS DMA, double-buffered.
// All LDS accesses via explicit AS3 pointers + integer offsets -> ds_load_b128.
// ---------------------------------------------------------------------------
__global__ __launch_bounds__(256)
void gemm_kernel(const _Float16* __restrict__ A, const _Float16* __restrict__ Wt,
                 const float* __restrict__ bias, const float* __restrict__ residual,
                 float* __restrict__ out)
{
    __shared__ char smem_raw[4 * TILE_BYTES];  // [bufA0|bufB0|bufA1|bufB1], 40 KB
    lchar lds = (lchar)smem_raw;

    const int t    = threadIdx.x;
    const int wid  = t >> 5;
    const int lane = t & 31;
    const int waveM = wid & 1;            // 0..1 -> 64 rows each
    const int waveN = wid >> 1;           // 0..3 -> 32 cols each
    const int halfSel = lane >> 4;
    const int l16 = lane & 15;

    // staging: 512 x 16B chunks per tile, 2 per thread per tile
    const int r0 = t >> 2;                // tile-local row of chunk 0 (0..63)
    const int r1 = r0 + 64;               // chunk 1 row (64..127)
    const int o0 = (t & 3) * 16;          // byte offset within 64B row (same for both)
    const char* Ag = (const char*)A + ((size_t)blockIdx.x * BM) * (DIM * 2) + o0;
    const char* Bg = (const char*)Wt + ((size_t)blockIdx.y * BN) * (DIM * 2) + o0;

    auto issue = [&](int bufOff, int kk) {   // kk in elements
        const size_t kB = (size_t)kk * 2;
        async_copy16(lds + bufOff + r0 * LPITCH + o0,
                     Ag + (size_t)r0 * (DIM * 2) + kB);
        async_copy16(lds + bufOff + r1 * LPITCH + o0,
                     Ag + (size_t)r1 * (DIM * 2) + kB);
        async_copy16(lds + bufOff + TILE_BYTES + r0 * LPITCH + o0,
                     Bg + (size_t)r0 * (DIM * 2) + kB);
        async_copy16(lds + bufOff + TILE_BYTES + r1 * LPITCH + o0,
                     Bg + (size_t)r1 * (DIM * 2) + kB);
    };

    v8f acc[4][2];
#pragma unroll
    for (int mi = 0; mi < 4; ++mi)
#pragma unroll
        for (int ni = 0; ni < 2; ++ni)
            acc[mi][ni] = (v8f){0.f, 0.f, 0.f, 0.f, 0.f, 0.f, 0.f, 0.f};

    const int mRow = waveM * 64;
    const int nRow = waveN * 32;
    // A 16x32 f16 frag: lanes 0-15 K[0..7],[16..23]; lanes 16-31 K[8..15],[24..31]
    const int aByte = halfSel ? 16 : 0;
    // B 32x16 frag: lanes 0-15 K[0..7],[16..23]; lanes 16-31 K[8..15],[24..31]
    const int bByte = halfSel ? 16 : 0;

    issue(0, 0);

    for (int s = 0; s < NK; ++s) {
        const int bufOff = (s & 1) * (2 * TILE_BYTES);
        const int nxtOff = ((s + 1) & 1) * (2 * TILE_BYTES);
        wait_async();          // my prefetch into buf[cur] is in LDS
        __syncthreads();       // everyone's is; everyone done reading the other buf
        if (s + 1 < NK) issue(nxtOff, (s + 1) * BK);

        union { v4i u[2]; v16h f; } af[4], bf[2];
#pragma unroll
        for (int mi = 0; mi < 4; ++mi) {
            const int ap = bufOff + (mRow + mi * 16 + l16) * LPITCH + aByte;
            af[mi].u[0] = *(lcv4)(lds + ap);
            af[mi].u[1] = *(lcv4)(lds + ap + 32);
        }
#pragma unroll
        for (int ni = 0; ni < 2; ++ni) {
            const int bp = bufOff + TILE_BYTES + (nRow + ni * 16 + l16) * LPITCH + bByte;
            bf[ni].u[0] = *(lcv4)(lds + bp);
            bf[ni].u[1] = *(lcv4)(lds + bp + 32);
        }
#pragma unroll
        for (int mi = 0; mi < 4; ++mi)
#pragma unroll
            for (int ni = 0; ni < 2; ++ni)
                acc[mi][ni] = __builtin_amdgcn_wmma_f32_16x16x32_f16(
                    false, af[mi].f, false, bf[ni].f,
                    (short)0, acc[mi][ni], false, false);
    }

    const int mBase = blockIdx.x * BM + mRow;
    const int nBase = blockIdx.y * BN + nRow;
#pragma unroll
    for (int mi = 0; mi < 4; ++mi) {
#pragma unroll
        for (int ni = 0; ni < 2; ++ni) {
            const int col = nBase + ni * 16 + l16;
            const float bval = bias[col];
#pragma unroll
            for (int i = 0; i < 8; ++i) {
                const int rrow = mBase + mi * 16 + i + 8 * halfSel;
                float val = acc[mi][ni][i] + bval;
                if (residual) val += residual[(size_t)rrow * DIM + col];
                out[(size_t)rrow * DIM + col] = val;
            }
        }
    }
}

extern "C" void kernel_launch(void* const* d_in, const int* in_sizes, int n_in,
                              void* d_out, int out_size, void* d_ws, size_t ws_size,
                              hipStream_t stream)
{
    const float* x    = (const float*)d_in[0];
    const float* v    = (const float*)d_in[1];
    const float* force= (const float*)d_in[2];
    const float* ln_g = (const float*)d_in[3];
    const float* ln_b = (const float*)d_in[4];
    const float* gW1  = (const float*)d_in[5];
    const float* gb1  = (const float*)d_in[6];
    const float* gW2  = (const float*)d_in[7];
    const float* gb2  = (const float*)d_in[8];
    const float* dtp  = (const float*)d_in[9];
    const float* cWx  = (const float*)d_in[10];
    const float* cWv  = (const float*)d_in[11];
    const float* cWo  = (const float*)d_in[12];
    const float* pxW  = (const float*)d_in[13];
    const float* pxb  = (const float*)d_in[14];
    const float* pvW  = (const float*)d_in[15];
    const float* pvb  = (const float*)d_in[16];

    const int B = in_sizes[0] / DIM;    // 8192

    _Float16* Xcat = (_Float16*)d_ws;
    _Float16* Vcat = Xcat + (size_t)B * DIM;
    _Float16* pxWt = Vcat + (size_t)B * DIM;
    _Float16* pvWt = pxWt + (size_t)DIM * DIM;

    float* out_x = (float*)d_out;
    float* out_v = out_x + (size_t)B * DIM;

    physics_kernel<<<B, 256, 0, stream>>>(x, v, force, ln_g, ln_b,
                                          gW1, gb1, gW2, gb2, dtp,
                                          cWx, cWv, cWo, Xcat, Vcat);

    dim3 tb(32, 8);
    convert_w<<<dim3(DIM / 32, DIM / 32), tb, 0, stream>>>(pxW, pxWt);
    convert_w<<<dim3(DIM / 32, DIM / 32), tb, 0, stream>>>(pvW, pvWt);

    gemm_kernel<<<dim3(B / 128, DIM / 128), 256, 0, stream>>>(Xcat, pxWt, pxb, x, out_x);
    gemm_kernel<<<dim3(B / 128, DIM / 128), 256, 0, stream>>>(Vcat, pvWt, pvb, nullptr, out_v);
}
